// BetterGCNLayer_62354335204074
// MI455X (gfx1250) — matrix operand order repaired and checked by reading, not compiled, not run on hardware
//
#include <hip/hip_runtime.h>
#include <hip/hip_bf16.h>
#include <math.h>

typedef _Float16 half_t;
typedef __attribute__((ext_vector_type(16))) _Float16 v16h;
typedef __attribute__((ext_vector_type(8)))  _Float16 v8h;
typedef __attribute__((ext_vector_type(8)))  float    v8f;

#define DIM 128

__device__ __forceinline__ void atomic_add_f32(float* p, float v) {
    __hip_atomic_fetch_add(p, v, __ATOMIC_RELAXED, __HIP_MEMORY_SCOPE_AGENT);
}

// ---------------------------------------------------------------- zero init
__global__ void gcn_zero_kernel(float* p, long n) {
    long i = (long)blockIdx.x * blockDim.x + threadIdx.x;
    long stride = (long)gridDim.x * blockDim.x;
    for (; i < n; i += stride) p[i] = 0.0f;
}

// ------------------------------------------------- W -> WMMA B-fragment swizzle
// B layout for V_WMMA_F32_16X16X32_F16 (from ISA 05_wmma B 64x16 pattern scaled
// to K=32): lane L holds column n = L%16; half j holds k = (L<16 ? 0 : 8) + (j&7) + 16*(j>>3).
// We need B[k][n] = W[o=n_global][k] (out = A @ W^T).
// Stored so each lane's 16 halfs are contiguous 32B:
//   wfrag[(((kc*8 + oc)*32 + L)*16 + j]
__global__ void gcn_prep_w_kernel(const float* __restrict__ W, half_t* __restrict__ wfrag) {
    int idx = blockIdx.x * blockDim.x + threadIdx.x;   // 0..16383
    int j  =  idx        & 15;
    int L  = (idx >> 4)  & 31;
    int oc = (idx >> 9)  & 7;
    int kc =  idx >> 12;
    int o = oc * 16 + (L & 15);
    int k = kc * 32 + ((L < 16) ? 0 : 8) + (j & 7) + ((j >> 3) << 4);
    wfrag[idx] = (half_t)W[o * DIM + k];
}

// ---------------------------------------------------------------- edge phase
// One wave32 per undirected edge record: lanes cover the 128 features as
// float4; scatter-add in both directions + degree counts. x and agg both live
// in the 192MB L2, so this is pure L2 atomic throughput.
__global__ __launch_bounds__(256) void gcn_edge_kernel(
        const float* __restrict__ x, const int* __restrict__ ei,
        float* __restrict__ agg, float* __restrict__ deg, int E) {
    long gw = ((long)blockIdx.x * blockDim.x + threadIdx.x) >> 5;
    int lane = threadIdx.x & 31;
    if (gw >= E) return;
    int s = ei[gw];
    int d = ei[(long)E + gw];
    float4 vs = ((const float4*)(x + (long)s * DIM))[lane];
    float4 vd = ((const float4*)(x + (long)d * DIM))[lane];
    float* ad = agg + (long)d * DIM + lane * 4;
    float* as = agg + (long)s * DIM + lane * 4;
    atomic_add_f32(ad + 0, vs.x); atomic_add_f32(ad + 1, vs.y);
    atomic_add_f32(ad + 2, vs.z); atomic_add_f32(ad + 3, vs.w);
    atomic_add_f32(as + 0, vd.x); atomic_add_f32(as + 1, vd.y);
    atomic_add_f32(as + 2, vd.z); atomic_add_f32(as + 3, vd.w);
    if (lane == 0) {
        atomic_add_f32(deg + d, 1.0f);
        atomic_add_f32(deg + s, 1.0f);
    }
}

// --------------------------------------- fused normalize+GEMM+bias+res+GELU
// Block = 256 threads = 8 waves, handles 16 nodes x 128 outputs.
// A-tile (16x128, deg-normalized, f16) staged in LDS with 136-half row pitch
// so the per-fragment ds_load_b128 (rows 0..15 at same column) is bank-
// conflict-free (row stride = 68 banks -> 4-bank rotation per row).
__global__ __launch_bounds__(256) void gcn_gemm_kernel(
        const float* __restrict__ agg, const float* __restrict__ deg,
        const half_t* __restrict__ wfrag, const float* __restrict__ bias,
        const float* __restrict__ x, float* __restrict__ out, int N) {
    __shared__ __attribute__((aligned(32))) half_t tileA[16][136];
    __shared__ float sscale[16];

    int t = threadIdx.x;
    long base = (long)blockIdx.x * 16;

    if (t < 16) {
        long n = base + t; if (n > N - 1) n = N - 1;
        float dg = deg[n];
        sscale[t] = rsqrtf(dg < 1.0f ? 1.0f : dg);   // 1/sqrt(max(deg,1))
    }
    __syncthreads();

    // stage normalized A tile as f16
    #pragma unroll
    for (int i = 0; i < 8; ++i) {
        int idx = i * 256 + t;             // 0..2047
        int row = idx >> 7, col = idx & 127;
        long n = base + row; if (n > N - 1) n = N - 1;
        tileA[row][col] = (half_t)(agg[n * DIM + col] * sscale[row]);
    }
    __syncthreads();

    int wv   = t >> 5;        // output column tile 0..7
    int lane = t & 31;
    int arow = lane & 15;
    int aoff = (lane < 16) ? 0 : 8;   // A-fragment K sub-block per ISA layout

    v8f c = {};
    #pragma unroll
    for (int kc = 0; kc < 4; ++kc) {
        // A: h[0..7]=K[kc*32+aoff .. +7], h[8..15]=K[kc*32+aoff+16 .. +7]
        v8h lo = *(const v8h*)&tileA[arow][kc * 32 + aoff];
        v8h hi = *(const v8h*)&tileA[arow][kc * 32 + aoff + 16];
        v16h a;
        #pragma unroll
        for (int j = 0; j < 8; ++j) { a[j] = lo[j]; a[j + 8] = hi[j]; }
        // B: pre-swizzled, contiguous 32B per lane
        v16h b = *(const v16h*)(wfrag + ((((kc * 8 + wv) * 32) + lane) << 4));
        c = __builtin_amdgcn_wmma_f32_16x16x32_f16(
                false, a, false, b, (short)0, c, false, false);
    }

    // C/D layout: VGPR r, lanes 0-15 -> M=r, lanes 16-31 -> M=8+r; N = lane%16
    int col   = wv * 16 + (lane & 15);
    int mbase = (lane < 16) ? 0 : 8;
    float bcol = bias[col];
    #pragma unroll
    for (int r = 0; r < 8; ++r) {
        long node = base + mbase + r;
        if (node < N) {
            float v = c[r] + bcol + x[node * DIM + col];   // bias + residual
            float g = 0.5f * v * (1.0f + erff(v * 0.70710678118654752f)); // exact GELU
            out[node * DIM + col] = g;
        }
    }
}

extern "C" void kernel_launch(void* const* d_in, const int* in_sizes, int n_in,
                              void* d_out, int out_size, void* d_ws, size_t ws_size,
                              hipStream_t stream) {
    const float* x    = (const float*)d_in[0];
    const int*   ei   = (const int*)  d_in[1];
    const float* W    = (const float*)d_in[2];
    const float* bias = (const float*)d_in[3];
    float*       out  = (float*)d_out;

    int N = in_sizes[0] / DIM;
    int E = in_sizes[1] / 2;

    // workspace: agg [N*128 f32] | deg [N f32] | wfrag [16384 f16]
    float*  agg   = (float*)d_ws;
    float*  deg   = agg + (long)N * DIM;
    half_t* wfrag = (half_t*)(deg + N);

    long nz = (long)N * DIM + N;
    gcn_zero_kernel<<<2048, 256, 0, stream>>>(agg, nz);
    gcn_prep_w_kernel<<<64, 256, 0, stream>>>(W, wfrag);

    long waves = E;                          // one wave per edge record
    int edgeBlocks = (int)((waves + 7) / 8); // 8 waves per 256-thread block
    gcn_edge_kernel<<<edgeBlocks, 256, 0, stream>>>(x, ei, agg, deg, E);

    int gemmBlocks = (N + 15) / 16;
    gcn_gemm_kernel<<<gemmBlocks, 256, 0, stream>>>(agg, deg, wfrag, bias, x, out, N);
}
